// NodeSchNetWrapper_19868518711759
// MI455X (gfx1250) — matrix operand order repaired and checked by reading, not compiled, not run on hardware
//
#include <hip/hip_runtime.h>

// ---------------------------------------------------------------------------
// SchNet forward on MI455X (gfx1250): bf16 WMMA GEMMs with fused epilogues.
// Fragments built via vectorized 16B LDS loads + bit_cast (no b16 packing).
// ---------------------------------------------------------------------------

#define Nn   10000
#define Ee   64000
#define Bb   128
#define Hh   600
#define Gg   50
#define Ll   6
#define NP   640          // padded hidden (mult of 32)
#define GP   64           // padded gaussians
#define LOG2F_ 0.69314718056f

typedef __attribute__((ext_vector_type(16))) __bf16 v16bf;
typedef __attribute__((ext_vector_type(8)))  float  v8f;

struct Pair128 { uint4 a, b; };   // 32 bytes == one v16bf fragment

// ---- bf16 helpers (RNE) ----------------------------------------------------
__device__ __forceinline__ unsigned short f2bf(float f) {
    unsigned int u = __float_as_uint(f);
    unsigned int r = u + 0x7FFFu + ((u >> 16) & 1u);
    return (unsigned short)(r >> 16);
}
__device__ __forceinline__ float bf2f(unsigned short u) {
    return __uint_as_float(((unsigned int)u) << 16);
}
__device__ __forceinline__ float sspf(float x) {   // softplus(x) - log(2)
    float sp = fmaxf(x, 0.f) + log1pf(__expf(-fabsf(x)));
    return sp - LOG2F_;
}
// two aligned 16B LDS reads -> one 32B bf16 fragment
__device__ __forceinline__ v16bf ld_frag(const unsigned short* p0,
                                         const unsigned short* p1) {
    Pair128 t;
    t.a = *(const uint4*)p0;
    t.b = *(const uint4*)p1;
    return __builtin_bit_cast(v16bf, t);
}

// ---- epilogue modes --------------------------------------------------------
enum { EPI_BF16 = 0, EPI_SSP_BF16 = 1, EPI_SCATTER = 2, EPI_RESID = 3, EPI_OUT = 4 };

// ---------------------------------------------------------------------------
// Tiled bf16 WMMA GEMM:  D[M,NP] = epi(A[M,K](bf16) @ Bm[K,NP](bf16) + bias)
// block = 256 threads (8 waves), 64x64 output tile, K stepped by 32 via LDS.
// A staged row-major [64][32]; B staged TRANSPOSED [64 n][32 k] so that both
// WMMA operand fragments are contiguous 16B runs in LDS.
// ---------------------------------------------------------------------------
template <int EPI>
__global__ __launch_bounds__(256) void gemm_bf16_wmma(
    const unsigned short* __restrict__ A,
    const unsigned short* __restrict__ Bm,
    const float* __restrict__ bias,
    int M, int K,
    unsigned short* __restrict__ outb,           // EPI_BF16 / EPI_SSP_BF16
    float* __restrict__ hf, unsigned short* __restrict__ hb,   // EPI_RESID
    const int* __restrict__ esrc, const int* __restrict__ edst, // EPI_SCATTER
    const float* __restrict__ Ccut, const unsigned short* __restrict__ yb,
    float* __restrict__ agg,
    float* __restrict__ outf, int outN)          // EPI_OUT
{
    __shared__ __align__(16) unsigned short sA[64 * 32];   // [row m][k]
    __shared__ __align__(16) unsigned short sBt[64 * 32];  // [col n][k]  (transposed)

    const int tid  = threadIdx.x;
    const int lane = tid & 31;
    const int wave = tid >> 5;
    const int half = lane >> 4;
    const int lcol = lane & 15;
    const int wm   = wave >> 1;          // 0..3  (M sub-tile)
    const int wn0  = (wave & 1) * 2;     // 0 or 2 (first of two N sub-tiles)
    const int m0   = blockIdx.y * 64;
    const int n0   = blockIdx.x * 64;

    v8f acc0 = {};
    v8f acc1 = {};

    // per-lane fragment base pointers (constant across K loop)
    const unsigned short* aBase  = sA  + (wm * 16 + lcol) * 32 + 8 * half;
    const unsigned short* b0Base = sBt + ((wn0 + 0) * 16 + lcol) * 32 + 8 * half;
    const unsigned short* b1Base = sBt + ((wn0 + 1) * 16 + lcol) * 32 + 8 * half;

    for (int k0 = 0; k0 < K; k0 += 32) {
        __syncthreads();
        {   // stage A tile 64x32 (row-major, one b128 store per thread)
            int row = tid >> 2, cg = (tid & 3) * 8;
            int gm  = m0 + row;
            uint4 v = {0u, 0u, 0u, 0u};
            if (gm < M)
                v = *(const uint4*)(A + (size_t)gm * K + (size_t)(k0 + cg));
            *(uint4*)(sA + row * 32 + cg) = v;
        }
        {   // stage B tile 32x64 -> transposed LDS [n][k]
            int row = tid >> 3, cn = (tid & 7) * 8;   // row = k (0..31), cn = n base
            Pair128 raw;
            raw.a = *(const uint4*)(Bm + (size_t)(k0 + row) * NP + (size_t)(n0 + cn));
            const unsigned short* e = (const unsigned short*)&raw.a;
            #pragma unroll
            for (int q = 0; q < 8; ++q)
                sBt[(cn + q) * 32 + row] = e[q];
        }
        if (k0 + 32 < K)   // hint next B tile toward L2/L0 (global_prefetch_b8)
            __builtin_prefetch(Bm + (size_t)(k0 + 32) * NP + (size_t)(n0 + (tid & 7) * 8), 0, 1);
        __syncthreads();

        // A fragment: elems [0..7] k=8*half+i ; [8..15] k=16+8*half+i  (contiguous)
        v16bf af  = ld_frag(aBase, aBase + 16);
        // B fragments: fixed column, elems [0..7] k=8*half+i ; [8..15] k=16+8*half+i
        v16bf bf0 = ld_frag(b0Base, b0Base + 16);
        v16bf bf1 = ld_frag(b1Base, b1Base + 16);

        acc0 = __builtin_amdgcn_wmma_f32_16x16x32_bf16(false, af, false, bf0,
                                                       (short)0, acc0, false, false);
        acc1 = __builtin_amdgcn_wmma_f32_16x16x32_bf16(false, af, false, bf1,
                                                       (short)0, acc1, false, false);
    }

    // ---- epilogue (C/D layout: m = i + 8*half, n = lane&15) ----
    #pragma unroll
    for (int j = 0; j < 2; ++j) {
        const v8f& acc = j ? acc1 : acc0;
        const int n = n0 + (wn0 + j) * 16 + lcol;
        #pragma unroll
        for (int i = 0; i < 8; ++i) {
            int m = m0 + wm * 16 + i + 8 * half;
            if (m >= M) continue;
            float val = acc[i] + (bias ? bias[n] : 0.f);
            if (EPI == EPI_BF16) {
                outb[(size_t)m * NP + n] = f2bf(val);
            } else if (EPI == EPI_SSP_BF16) {
                outb[(size_t)m * NP + n] = f2bf(sspf(val));
            } else if (EPI == EPI_SCATTER) {
                // m is an edge: W = val*C[e]; msg = W * xj[src[e],n]; agg[dst[e],n]+=msg
                float w  = val * Ccut[m];
                int   s  = esrc[m];
                int   t  = edst[m];
                float xj = bf2f(yb[(size_t)s * NP + n]);
                atomicAdd(agg + (size_t)t * NP + n, w * xj);
            } else if (EPI == EPI_RESID) {
                size_t off = (size_t)m * NP + n;
                float nh = hf[off] + val;
                hf[off] = nh;
                hb[off] = f2bf(nh);
            } else { // EPI_OUT
                if (n < outN) outf[(size_t)m * outN + n] = val;
            }
        }
    }
}

// ---------------------------------------------------------------------------
// Small elementwise kernels
// ---------------------------------------------------------------------------
__global__ void pad_w_bf16(const float* __restrict__ src, unsigned short* __restrict__ dst,
                           int M, int N, int Mp, int Np_) {
    int z = blockIdx.z;
    src += (size_t)z * M * N;
    dst += (size_t)z * Mp * Np_;
    int t = blockIdx.x * blockDim.x + threadIdx.x;
    if (t >= Mp * Np_) return;
    int m = t / Np_, n = t % Np_;
    float v = (m < M && n < N) ? src[(size_t)m * N + n] : 0.f;
    dst[t] = f2bf(v);
}

__global__ void pad_bias_f32(const float* __restrict__ src, float* __restrict__ dst,
                             int N, int Np_) {
    int z = blockIdx.z;
    src += (size_t)z * N;
    dst += (size_t)z * Np_;
    int t = blockIdx.x * blockDim.x + threadIdx.x;
    if (t >= Np_) return;
    dst[t] = (t < N) ? src[t] : 0.f;
}

__global__ void edge_geom_k(const int* __restrict__ src, const int* __restrict__ dst,
                            const float* __restrict__ pos,
                            unsigned short* __restrict__ rbf, float* __restrict__ Cc) {
    int e = blockIdx.x * blockDim.x + threadIdx.x;
    if (e >= Ee) return;
    int s = src[e], d = dst[e];
    float dx = pos[3 * s + 0] - pos[3 * d + 0];
    float dy = pos[3 * s + 1] - pos[3 * d + 1];
    float dz = pos[3 * s + 2] - pos[3 * d + 2];
    float dist = sqrtf(dx * dx + dy * dy + dz * dz + 1e-12f);
    const float delta = 10.0f / 49.0f;
    const float coeff = -0.5f / (delta * delta);
    #pragma unroll 8
    for (int g = 0; g < GP; ++g) {
        float v = 0.f;
        if (g < Gg) { float t = dist - g * delta; v = __expf(coeff * t * t); }
        rbf[(size_t)e * GP + g] = f2bf(v);
    }
    Cc[e] = 0.5f * (__cosf(dist * 0.31415926535f) + 1.0f);
}

__global__ void embed_k(const int* __restrict__ z, const float* __restrict__ emb,
                        float* __restrict__ hf, unsigned short* __restrict__ hb) {
    size_t t = (size_t)blockIdx.x * blockDim.x + threadIdx.x;
    if (t >= (size_t)Nn * NP) return;
    int i = (int)(t / NP), c = (int)(t % NP);
    float v = (c < Hh) ? emb[(size_t)z[i] * Hh + c] : 0.f;
    hf[t] = v;
    hb[t] = f2bf(v);
}

__global__ void cvt_f32_bf16_k(const float* __restrict__ s, unsigned short* __restrict__ d,
                               size_t n) {
    size_t t = (size_t)blockIdx.x * blockDim.x + threadIdx.x;
    if (t < n) d[t] = f2bf(s[t]);
}

__global__ void pool_sum_k(const int* __restrict__ batch, const float* __restrict__ hf,
                           float* __restrict__ pooled, float* __restrict__ cnt) {
    int i = blockIdx.x;                 // atom
    int b = batch[i];
    for (int c = threadIdx.x; c < NP; c += blockDim.x)
        atomicAdd(pooled + (size_t)b * NP + c, hf[(size_t)i * NP + c]);
    if (threadIdx.x == 0) atomicAdd(cnt + b, 1.0f);
}

__global__ void pool_mean_k(const float* __restrict__ pooled, const float* __restrict__ cnt,
                            unsigned short* __restrict__ pooledb) {
    int t = blockIdx.x * blockDim.x + threadIdx.x;
    if (t >= Bb * NP) return;
    int b = t / NP;
    pooledb[t] = f2bf(pooled[t] / fmaxf(cnt[b], 1.0f));
}

// ---------------------------------------------------------------------------
// Host launcher
// ---------------------------------------------------------------------------
extern "C" void kernel_launch(void* const* d_in, const int* in_sizes, int n_in,
                              void* d_out, int out_size, void* d_ws, size_t ws_size,
                              hipStream_t stream) {
    (void)in_sizes; (void)n_in; (void)out_size; (void)ws_size;

    const int*   z       = (const int*)  d_in[0];
    const float* pos     = (const float*)d_in[1];
    const int*   batch   = (const int*)  d_in[2];
    const int*   eidx    = (const int*)  d_in[3];
    const float* emb     = (const float*)d_in[4];
    const float* mlp_w1  = (const float*)d_in[5];
    const float* mlp_b1  = (const float*)d_in[6];
    const float* mlp_w2  = (const float*)d_in[7];
    const float* mlp_b2  = (const float*)d_in[8];
    const float* lin1_w  = (const float*)d_in[9];
    const float* lin2_w  = (const float*)d_in[10];
    const float* lin2_b  = (const float*)d_in[11];
    const float* intw    = (const float*)d_in[12];
    const float* intb    = (const float*)d_in[13];
    const float* pool_w  = (const float*)d_in[14];
    const float* pool_b  = (const float*)d_in[15];
    float* out = (float*)d_out;

    const int* esrc = eidx;
    const int* edst = eidx + Ee;

    // ---- workspace carve-out (256B aligned) ----
    char* base = (char*)d_ws;
    size_t cur = 0;
    auto alloc = [&](size_t bytes) -> char* {
        char* p = base + cur;
        cur += (bytes + 255) & ~(size_t)255;
        return p;
    };
    unsigned short* rbf_b  = (unsigned short*)alloc((size_t)Ee * GP * 2);
    float*          Cc     = (float*)         alloc((size_t)Ee * 4);
    float*          hf     = (float*)         alloc((size_t)Nn * NP * 4);
    unsigned short* hb     = (unsigned short*)alloc((size_t)Nn * NP * 2);
    unsigned short* yb     = (unsigned short*)alloc((size_t)Nn * NP * 2);
    float*          aggf   = (float*)         alloc((size_t)Nn * NP * 4);
    unsigned short* aggb   = (unsigned short*)alloc((size_t)Nn * NP * 2);
    unsigned short* t1b    = (unsigned short*)alloc((size_t)Nn * NP * 2);
    unsigned short* tmpb   = (unsigned short*)alloc((size_t)Ee * NP * 2);
    unsigned short* w1p    = (unsigned short*)alloc((size_t)Ll * GP * NP * 2);
    unsigned short* w2p    = (unsigned short*)alloc((size_t)Ll * NP * NP * 2);
    unsigned short* lin1p  = (unsigned short*)alloc((size_t)Ll * NP * NP * 2);
    unsigned short* lin2p  = (unsigned short*)alloc((size_t)Ll * NP * NP * 2);
    unsigned short* intp   = (unsigned short*)alloc((size_t)Ll * NP * NP * 2);
    unsigned short* poolp  = (unsigned short*)alloc((size_t)NP * NP * 2);
    float*          b1p    = (float*)alloc((size_t)Ll * NP * 4);
    float*          b2p    = (float*)alloc((size_t)Ll * NP * 4);
    float*          l2bp   = (float*)alloc((size_t)Ll * NP * 4);
    float*          ibp    = (float*)alloc((size_t)Ll * NP * 4);
    float*          pbp    = (float*)alloc((size_t)NP * 4);
    float*          pooled = (float*)alloc((size_t)Bb * NP * 4);
    float*          cntv   = (float*)alloc((size_t)Bb * 4);
    unsigned short* pooledb= (unsigned short*)alloc((size_t)Bb * NP * 2);

    // ---- 1) pad weights/biases to bf16 / padded f32 ----
    {
        dim3 blk(256);
        dim3 gW((NP * NP + 255) / 256, 1, Ll);
        dim3 g1((GP * NP + 255) / 256, 1, Ll);
        dim3 gP((NP * NP + 255) / 256, 1, 1);
        pad_w_bf16<<<g1, blk, 0, stream>>>(mlp_w1, w1p, Gg, Hh, GP, NP);
        pad_w_bf16<<<gW, blk, 0, stream>>>(mlp_w2, w2p, Hh, Hh, NP, NP);
        pad_w_bf16<<<gW, blk, 0, stream>>>(lin1_w, lin1p, Hh, Hh, NP, NP);
        pad_w_bf16<<<gW, blk, 0, stream>>>(lin2_w, lin2p, Hh, Hh, NP, NP);
        pad_w_bf16<<<gW, blk, 0, stream>>>(intw,   intp,  Hh, Hh, NP, NP);
        pad_w_bf16<<<gP, blk, 0, stream>>>(pool_w, poolp, Hh, Hh, NP, NP);
        dim3 gb((NP + 255) / 256, 1, Ll);
        dim3 gb1((NP + 255) / 256, 1, 1);
        pad_bias_f32<<<gb,  blk, 0, stream>>>(mlp_b1, b1p,  Hh, NP);
        pad_bias_f32<<<gb,  blk, 0, stream>>>(mlp_b2, b2p,  Hh, NP);
        pad_bias_f32<<<gb,  blk, 0, stream>>>(lin2_b, l2bp, Hh, NP);
        pad_bias_f32<<<gb,  blk, 0, stream>>>(intb,   ibp,  Hh, NP);
        pad_bias_f32<<<gb1, blk, 0, stream>>>(pool_b, pbp,  Hh, NP);
    }

    // ---- 2) edge geometry + embedding ----
    edge_geom_k<<<(Ee + 255) / 256, 256, 0, stream>>>(esrc, edst, pos, rbf_b, Cc);
    embed_k<<<(int)(((size_t)Nn * NP + 255) / 256), 256, 0, stream>>>(z, emb, hf, hb);

    const dim3 gblk(256);
    const dim3 gridE(NP / 64, (Ee + 63) / 64);
    const dim3 gridN(NP / 64, (Nn + 63) / 64);
    const dim3 gridB(NP / 64, (Bb + 63) / 64);

    // ---- 3) interaction layers ----
    for (int k = 0; k < Ll; ++k) {
        const unsigned short* w1k = w1p  + (size_t)k * GP * NP;
        const unsigned short* w2k = w2p  + (size_t)k * NP * NP;
        const unsigned short* l1k = lin1p + (size_t)k * NP * NP;
        const unsigned short* l2k = lin2p + (size_t)k * NP * NP;
        const unsigned short* ik  = intp  + (size_t)k * NP * NP;
        const float* b1k = b1p  + (size_t)k * NP;
        const float* b2k = b2p  + (size_t)k * NP;
        const float* l2b = l2bp + (size_t)k * NP;
        const float* ib  = ibp  + (size_t)k * NP;

        // tmp = ssp(rbf @ w1 + b1)            [E,NP] bf16
        gemm_bf16_wmma<EPI_SSP_BF16><<<gridE, gblk, 0, stream>>>(
            rbf_b, w1k, b1k, Ee, GP, tmpb,
            nullptr, nullptr, nullptr, nullptr, nullptr, nullptr, nullptr,
            nullptr, 0);
        // y = h @ lin1                         [N,NP] bf16
        gemm_bf16_wmma<EPI_BF16><<<gridN, gblk, 0, stream>>>(
            hb, l1k, nullptr, Nn, NP, yb,
            nullptr, nullptr, nullptr, nullptr, nullptr, nullptr, nullptr,
            nullptr, 0);
        // agg = scatter_add(dst, (tmp@w2+b2)*C * y[src])
        hipMemsetAsync(aggf, 0, (size_t)Nn * NP * 4, stream);
        gemm_bf16_wmma<EPI_SCATTER><<<gridE, gblk, 0, stream>>>(
            tmpb, w2k, b2k, Ee, NP, nullptr,
            nullptr, nullptr, esrc, edst, Cc, yb, aggf,
            nullptr, 0);
        cvt_f32_bf16_k<<<(int)(((size_t)Nn * NP + 255) / 256), 256, 0, stream>>>(
            aggf, aggb, (size_t)Nn * NP);
        // t1 = ssp(agg @ lin2 + b)             [N,NP] bf16
        gemm_bf16_wmma<EPI_SSP_BF16><<<gridN, gblk, 0, stream>>>(
            aggb, l2k, l2b, Nn, NP, t1b,
            nullptr, nullptr, nullptr, nullptr, nullptr, nullptr, nullptr,
            nullptr, 0);
        // h += t1 @ int_lin + b  (residual, refresh bf16 copy)
        gemm_bf16_wmma<EPI_RESID><<<gridN, gblk, 0, stream>>>(
            t1b, ik, ib, Nn, NP, nullptr,
            hf, hb, nullptr, nullptr, nullptr, nullptr, nullptr,
            nullptr, 0);
    }

    // ---- 4) segment-mean pooling ----
    hipMemsetAsync(pooled, 0, (size_t)Bb * NP * 4, stream);
    hipMemsetAsync(cntv,   0, (size_t)Bb * 4,      stream);
    pool_sum_k<<<Nn, 256, 0, stream>>>(batch, hf, pooled, cntv);
    pool_mean_k<<<(Bb * NP + 255) / 256, 256, 0, stream>>>(pooled, cntv, pooledb);

    // ---- 5) out = pooled @ pool_w + pool_b   [B,600] f32 ----
    gemm_bf16_wmma<EPI_OUT><<<gridB, gblk, 0, stream>>>(
        pooledb, poolp, pbp, Bb, NP, nullptr,
        nullptr, nullptr, nullptr, nullptr, nullptr, nullptr, nullptr,
        out, Hh);
}
